// AttentionHead_18949395710288
// MI455X (gfx1250) — matrix operand order repaired and checked by reading, not compiled, not run on hardware
//
#include <hip/hip_runtime.h>

typedef __attribute__((ext_vector_type(16))) _Float16 v16h;
typedef __attribute__((ext_vector_type(8)))  _Float16 v8h;
typedef __attribute__((ext_vector_type(8)))  float    v8f;

#define BB   8
#define TT   2048
#define CC   1024
#define HD   64
#define N3   192           // 3 * HD
#define MTOT (BB * TT)     // 16384 rows

__device__ inline v8f zero8() {
    v8f z = {0.f, 0.f, 0.f, 0.f, 0.f, 0.f, 0.f, 0.f};
    return z;
}

__device__ inline v16h cat8(v8h lo, v8h hi) {
    return __builtin_shufflevector(lo, hi, 0, 1, 2, 3, 4, 5, 6, 7,
                                   8, 9, 10, 11, 12, 13, 14, 15);
}

// ---------------------------------------------------------------------------
// Kernel 0: convert W (fp32, [C, 3*HD]) to f16, TRANSPOSED: Wt[n][k], n<192.
// ---------------------------------------------------------------------------
__global__ void cvt_w_kernel(const float* __restrict__ W, _Float16* __restrict__ Wt, int n) {
    int i = blockIdx.x * 256 + threadIdx.x;          // i = ncol * CC + k
    if (i < n) {
        int ncol = i / CC;
        int k    = i - ncol * CC;
        Wt[i] = (_Float16)W[(size_t)k * N3 + ncol];
    }
}

// ---------------------------------------------------------------------------
// Kernel 1: QKV projection  kqv = x @ W + b.
// K, Q row-major f16 [B*T, 64]; V transposed f16 Vt[b][d][t].
// ---------------------------------------------------------------------------
__global__ __launch_bounds__(256)
void qkv_gemm_kernel(const float* __restrict__ x, const _Float16* __restrict__ Wt,
                     const float* __restrict__ bias,
                     _Float16* __restrict__ Ko, _Float16* __restrict__ Qo,
                     _Float16* __restrict__ Vt) {
    __shared__ __align__(32) _Float16 Wlds[N3 * 32];   // 12 KB, [n][kk] layout

    const int tid  = threadIdx.x;
    const int wave = tid >> 5;
    const int lane = tid & 31;
    const int ln   = lane & 15;
    const int h    = lane >> 4;
    const int rowbase = blockIdx.x * 128 + wave * 16;

    v8f acc[12];
    #pragma unroll
    for (int t = 0; t < 12; ++t) acc[t] = zero8();

    const float*    xr_base = x + (size_t)(rowbase + ln) * CC + 16 * h;
    const _Float16* wp      = &Wlds[ln * 32 + 16 * h];   // fragment t at +t*512

    for (int kb = 0; kb < CC; kb += 32) {
        // Stage Wt[:, kb..kb+31] into LDS: one 64-byte row per thread.
        if (tid < N3) {
            const _Float16* src = Wt + (size_t)tid * CC + kb;
            v16h w0 = *(const v16h*)(src);
            v16h w1 = *(const v16h*)(src + 16);
            *(v16h*)&Wlds[tid * 32]      = w0;
            *(v16h*)&Wlds[tid * 32 + 16] = w1;
        }
        __syncthreads();

        // A fragment (16x32 f16): row M=ln, K = kb + e + 16h
        const float* xr = xr_base + kb;
        if (kb + 32 < CC) __builtin_prefetch(xr + 32, 0, 1);
        v8f xlo = *(const v8f*)(xr);
        v8f xhi = *(const v8f*)(xr + 8);
        v16h a;
        #pragma unroll
        for (int e = 0; e < 8; ++e) a[e]     = (_Float16)xlo[e];
        #pragma unroll
        for (int e = 0; e < 8; ++e) a[8 + e] = (_Float16)xhi[e];

        // Rotated software pipeline: issue load of fragment t+1 BEFORE the
        // WMMA on fragment t, so LDS latency hides behind the matrix op.
        v16h cur = *(const v16h*)(wp);
        #pragma unroll
        for (int t = 0; t < 11; ++t) {
            v16h nxt = *(const v16h*)(wp + (t + 1) * 512);
            acc[t] = __builtin_amdgcn_wmma_f32_16x16x32_f16(
                false, a, false, cur, (short)0, acc[t], false, false);
            cur = nxt;
        }
        acc[11] = __builtin_amdgcn_wmma_f32_16x16x32_f16(
            false, a, false, cur, (short)0, acc[11], false, false);
        __syncthreads();
    }

    // Epilogue: add bias; K,Q row-major; V transposed (dim-major).
    const int b_idx = rowbase / TT;                   // 128-row block never crosses batch
    const int tbase = rowbase - b_idx * TT;
    #pragma unroll
    for (int t = 0; t < 12; ++t) {
        const int ncol = t * 16 + ln;
        const float bv = bias[ncol];
        #pragma unroll
        for (int r = 0; r < 8; ++r) {
            const int row = rowbase + r + 8 * h;
            const _Float16 val = (_Float16)(acc[t][r] + bv);
            if (t < 4) {
                Ko[(size_t)row * HD + ncol] = val;
            } else if (t < 8) {
                Qo[(size_t)row * HD + (ncol - 64)] = val;
            } else {
                const int d = ncol - 128;
                Vt[((size_t)b_idx * HD + d) * TT + (tbase + r + 8 * h)] = val;
            }
        }
    }
}

// ---------------------------------------------------------------------------
// Kernel 2: causal flash attention, one wave per (16-query tile, batch).
// 64 keys per block: 8 S-WMMAs, 8 PV-WMMAs, one softmax update.
// ---------------------------------------------------------------------------
__global__ __launch_bounds__(32)
void attn_kernel(const _Float16* __restrict__ Qw, const _Float16* __restrict__ Kw,
                 const _Float16* __restrict__ Vt, float* __restrict__ out) {
    __shared__ __align__(32) _Float16 Plds[16 * 64];  // 2 KB P tile

    const int lane = threadIdx.x;
    const int ln   = lane & 15;
    const int h    = lane >> 4;
    const int b    = blockIdx.y;
    const int qbase = blockIdx.x * 16;

    const _Float16* Qb  = Qw + ((size_t)b * TT + qbase) * HD;
    const _Float16* Kb  = Kw + (size_t)b * TT * HD;
    const _Float16* Vtb = Vt + (size_t)b * HD * TT;

    // Q A-fragments for head-dim halves [0,32) and [32,64).
    v16h qa[2];
    #pragma unroll
    for (int hf = 0; hf < 2; ++hf) {
        const _Float16* qr = Qb + (size_t)ln * HD + hf * 32 + 16 * h;
        qa[hf] = cat8(*(const v8h*)(qr), *(const v8h*)(qr + 8));
    }

    v8f o[4];
    #pragma unroll
    for (int t = 0; t < 4; ++t) o[t] = zero8();
    float m[8], l[8];
    #pragma unroll
    for (int r = 0; r < 8; ++r) { m[r] = -1e30f; l[r] = 0.f; }

    for (int jb = 0; jb < qbase + 16; jb += 64) {     // 64 keys per block
        // --- Preload all 8 K fragments (single vmem clause) ---
        v16h kf[8];
        #pragma unroll
        for (int kt = 0; kt < 4; ++kt)
            #pragma unroll
            for (int hf = 0; hf < 2; ++hf)
                kf[kt * 2 + hf] = *(const v16h*)(
                    Kb + (size_t)(jb + kt * 16 + ln) * HD + hf * 32 + 16 * h);

        // --- S = Q @ K^T (four 16-key sub-tiles) ---
        v8f s[4];
        #pragma unroll
        for (int kt = 0; kt < 4; ++kt) {
            v8f st = zero8();
            #pragma unroll
            for (int hf = 0; hf < 2; ++hf)
                st = __builtin_amdgcn_wmma_f32_16x16x32_f16(
                    false, qa[hf], false, kf[kt * 2 + hf], (short)0, st, false, false);
            s[kt] = st;
        }

        // --- scale + causal mask (M = r + 8h, N = ln) ---
        #pragma unroll
        for (int kt = 0; kt < 4; ++kt)
            #pragma unroll
            for (int r = 0; r < 8; ++r) {
                const int qi = qbase + r + 8 * h;
                const int ki = jb + kt * 16 + ln;
                const float v = s[kt][r] * 0.125f;    // HEAD_DIM^-0.5
                s[kt][r] = (ki <= qi) ? v : -1e30f;
            }

        // --- online softmax (row reductions across the 16-lane group) ---
        float corr[8];
        #pragma unroll
        for (int r = 0; r < 8; ++r) {
            float mx = fmaxf(fmaxf(s[0][r], s[1][r]), fmaxf(s[2][r], s[3][r]));
            #pragma unroll
            for (int off = 1; off < 16; off <<= 1)
                mx = fmaxf(mx, __shfl_xor(mx, off, 32));
            const float mn = fmaxf(m[r], mx);
            const float c  = __expf(m[r] - mn);
            float rs = 0.f;
            #pragma unroll
            for (int kt = 0; kt < 4; ++kt) {
                const float p = __expf(s[kt][r] - mn);
                s[kt][r] = p;
                rs += p;
            }
            #pragma unroll
            for (int off = 1; off < 16; off <<= 1)
                rs += __shfl_xor(rs, off, 32);
            l[r] = l[r] * c + rs;
            m[r] = mn;
            corr[r] = c;
        }

        #pragma unroll
        for (int t = 0; t < 4; ++t)
            #pragma unroll
            for (int r = 0; r < 8; ++r) o[t][r] *= corr[r];

        // --- P (C-layout f32) -> LDS (f16, row-major, stride 64) ---
        #pragma unroll
        for (int kt = 0; kt < 4; ++kt)
            #pragma unroll
            for (int r = 0; r < 8; ++r)
                Plds[(r + 8 * h) * 64 + kt * 16 + ln] = (_Float16)s[kt][r];
        __syncthreads();

        // Two A-layout P fragments (keys [0,32) and [32,64) of this block)
        v16h pa[2];
        #pragma unroll
        for (int f = 0; f < 2; ++f) {
            const _Float16* pr = Plds + ln * 64 + f * 32 + 16 * h;
            pa[f] = cat8(*(const v8h*)(pr), *(const v8h*)(pr + 8));
        }
        __syncthreads();

        // --- Preload all 8 V fragments, then 8 PV WMMAs ---
        v16h vf[8];
        #pragma unroll
        for (int t = 0; t < 4; ++t)
            #pragma unroll
            for (int f = 0; f < 2; ++f)
                vf[t * 2 + f] = *(const v16h*)(
                    Vtb + (size_t)(t * 16 + ln) * TT + jb + f * 32 + 16 * h);

        #pragma unroll
        for (int t = 0; t < 4; ++t)
            #pragma unroll
            for (int f = 0; f < 2; ++f)
                o[t] = __builtin_amdgcn_wmma_f32_16x16x32_f16(
                    false, pa[f], false, vf[t * 2 + f], (short)0, o[t], false, false);
    }

    // --- write normalized output (fp32), reciprocal instead of divide ---
    float rl[8];
    #pragma unroll
    for (int r = 0; r < 8; ++r) rl[r] = __frcp_rn(l[r]);
    float* ob = out + ((size_t)b * TT + qbase) * HD;
    #pragma unroll
    for (int t = 0; t < 4; ++t)
        #pragma unroll
        for (int r = 0; r < 8; ++r)
            ob[(size_t)(r + 8 * h) * HD + t * 16 + ln] = o[t][r] * rl[r];
}

// ---------------------------------------------------------------------------
extern "C" void kernel_launch(void* const* d_in, const int* in_sizes, int n_in,
                              void* d_out, int out_size, void* d_ws, size_t ws_size,
                              hipStream_t stream) {
    const float* x    = (const float*)d_in[0];   // [B, T, C]
    const float* W    = (const float*)d_in[1];   // [C, 3*HD]
    const float* bias = (const float*)d_in[2];   // [3*HD]
    float* out = (float*)d_out;                  // [B, T, HD] fp32

    // workspace layout (f16): K | Q | Vt | Wt
    _Float16* Kw = (_Float16*)d_ws;
    _Float16* Qw = Kw + (size_t)MTOT * HD;
    _Float16* Vt = Qw + (size_t)MTOT * HD;
    _Float16* Wt = Vt + (size_t)MTOT * HD;

    const int wn = CC * N3;
    cvt_w_kernel<<<(wn + 255) / 256, 256, 0, stream>>>(W, Wt, wn);

    qkv_gemm_kernel<<<MTOT / 128, 256, 0, stream>>>(x, Wt, bias, Kw, Qw, Vt);

    attn_kernel<<<dim3(TT / 16, BB), 32, 0, stream>>>(Qw, Kw, Vt, out);
}
